// BinaryLinear_21861383537186
// MI455X (gfx1250) — compile-verified, matrix operand order checked by reading, not run on hardware
//
#include <hip/hip_runtime.h>
#include <stdint.h>

// Problem constants (from reference)
#define IN_F   4096
#define OUT_F  4096
#define NTOK   8192

// Tiling
#define BM 128          // token rows per block
#define BN 128          // out-features per block
#define BK 32           // K slice (matches wmma 16x16x32 K)
#define LDT 40          // LDS row stride in halves (80B -> conflict-free b128 reads)

typedef __attribute__((ext_vector_type(16))) __bf16 v16bf;
typedef __attribute__((ext_vector_type(8)))  float  v8f;
typedef int v4i_ __attribute__((__vector_size__(16)));   // matches builtin param type

union FragU { uint4 q[2]; v16bf v; };

#if __has_builtin(__builtin_amdgcn_global_load_async_to_lds_b128)
#define HAVE_ASYNC_LDS 1
#else
#define HAVE_ASYNC_LDS 0
#endif

#if HAVE_ASYNC_LDS
__device__ __forceinline__ void async_copy_b128(const void* g, void* l) {
    // global (AS1) -> LDS (AS3), 16 bytes, tracked by ASYNCcnt
    __builtin_amdgcn_global_load_async_to_lds_b128(
        (__attribute__((address_space(1))) v4i_*)g,
        (__attribute__((address_space(3))) v4i_*)l,
        0, 0);
}
__device__ __forceinline__ void wait_async0() {
#if __has_builtin(__builtin_amdgcn_s_wait_asynccnt)
    __builtin_amdgcn_s_wait_asynccnt(0);
#else
    asm volatile("s_wait_asynccnt 0" ::: "memory");
#endif
}
#endif

__device__ __forceinline__ uint32_t pack2_bf16_rne(float a, float b) {
    uint32_t ua = __builtin_bit_cast(uint32_t, a);
    uint32_t ub = __builtin_bit_cast(uint32_t, b);
    ua += 0x7FFFu + ((ua >> 16) & 1u);   // round-to-nearest-even
    ub += 0x7FFFu + ((ub >> 16) & 1u);
    return (ua >> 16) | (ub & 0xFFFF0000u);
}

// ---------------------------------------------------------------------------
// Prologue 1: W fp32 -> bf16 {0.0, 1.0}  ((w > 0) ? 1 : 0 -- exact in bf16)
// ---------------------------------------------------------------------------
__global__ __launch_bounds__(256) void binarize_weights(
        const float* __restrict__ w, uint4* __restrict__ wb) {
    const int i = blockIdx.x * 256 + threadIdx.x;
    const float4* src = (const float4*)w + (size_t)i * 2;
    float4 f0 = src[0];
    float4 f1 = src[1];
    auto b1 = [](float x) -> uint32_t { return x > 0.0f ? 0x3F80u : 0u; };
    uint4 o;
    o.x = b1(f0.x) | (b1(f0.y) << 16);
    o.y = b1(f0.z) | (b1(f0.w) << 16);
    o.z = b1(f1.x) | (b1(f1.y) << 16);
    o.w = b1(f1.z) | (b1(f1.w) << 16);
    wb[i] = o;
}

// ---------------------------------------------------------------------------
// Prologue 2: X fp32 -> bf16 (RNE). Memory-bound one-shot pass; removes all
// conversion VALU from the K-loop and halves X read traffic in the GEMM.
// ---------------------------------------------------------------------------
__global__ __launch_bounds__(256) void convert_x(
        const float* __restrict__ x, uint4* __restrict__ xb) {
    const int i = blockIdx.x * 256 + threadIdx.x;
    const float4* src = (const float4*)x + (size_t)i * 2;
    float4 f0 = src[0];
    float4 f1 = src[1];
    uint4 o;
    o.x = pack2_bf16_rne(f0.x, f0.y);
    o.y = pack2_bf16_rne(f0.z, f0.w);
    o.z = pack2_bf16_rne(f1.x, f1.y);
    o.w = pack2_bf16_rne(f1.z, f1.w);
    xb[i] = o;
}

// ---------------------------------------------------------------------------
// Main GEMM: out = Xb * Wb^T + bias.
// 256 threads = 8 waves; wave grid 4(M) x 2(N); each wave: 32x64 output
// = 2x4 tiles of v_wmma_f32_16x16x32_bf16.
// Double-buffered LDS; async global->LDS staging overlapped with WMMAs;
// one barrier per K-step.
// XB16=true : A staged async from pre-converted bf16 X (fast path)
// XB16=false: A converted fp32->bf16 inline (ws too small fallback)
// ---------------------------------------------------------------------------
template <bool XB16>
__global__ __launch_bounds__(256) void binlinear_gemm(
        const float* __restrict__ X,           // fp32 (fallback path)
        const unsigned short* __restrict__ Xb, // bf16 bits (fast path)
        const unsigned short* __restrict__ Wb, // bf16 bits, [OUT_F][IN_F]
        const float* __restrict__ bias,
        float* __restrict__ Out) {
    __shared__ alignas(16) unsigned short sA[2][BM * LDT];
    __shared__ alignas(16) unsigned short sW[2][BN * LDT];

    const int tid  = threadIdx.x;
    const int lane = tid & 31;
    const int wave = tid >> 5;
    const int wm   = wave & 3;     // 0..3 -> 32-row slice
    const int wn   = wave >> 2;    // 0..1 -> 64-col slice
    const int lrow = lane & 15;    // matrix row/col index within 16
    const int h    = lane >> 4;    // lane half (K packing select)

    const int rowBase = blockIdx.x * BM;   // token rows
    const int colBase = blockIdx.y * BN;   // out features

    // staging coordinates: 2 threads per tile row, each moves 16 elements
    const int srow  = tid >> 1;    // 0..127
    const int shalf = tid & 1;     // which 16-wide half of the BK=32 slice

    v8f acc[2][4];
    #pragma unroll
    for (int i = 0; i < 2; ++i)
        #pragma unroll
        for (int j = 0; j < 4; ++j)
            acc[i][j] = {};

    const float*          xrow  = X  + (size_t)(rowBase + srow) * IN_F + shalf * 16;
    const unsigned short* xbrow = Xb + (size_t)(rowBase + srow) * IN_F + shalf * 16;
    const unsigned short* wrow  = Wb + (size_t)(colBase + srow) * IN_F + shalf * 16;

    auto stage = [&](int buf, int k0) {
        unsigned short* da = &sA[buf][srow * LDT + shalf * 16];
        unsigned short* dw = &sW[buf][srow * LDT + shalf * 16];
#if HAVE_ASYNC_LDS
        async_copy_b128(wrow + k0,     dw);
        async_copy_b128(wrow + k0 + 8, dw + 8);
        if constexpr (XB16) {
            async_copy_b128(xbrow + k0,     da);
            async_copy_b128(xbrow + k0 + 8, da + 8);
        }
#else
        {
            const uint4* ws = (const uint4*)(wrow + k0);
            uint4 w0 = ws[0];
            uint4 w1 = ws[1];
            ((uint4*)dw)[0] = w0;
            ((uint4*)dw)[1] = w1;
        }
        if constexpr (XB16) {
            const uint4* as = (const uint4*)(xbrow + k0);
            uint4 a0 = as[0];
            uint4 a1 = as[1];
            ((uint4*)da)[0] = a0;
            ((uint4*)da)[1] = a1;
        }
#endif
        if constexpr (!XB16) {
            const float4* xs = (const float4*)(xrow + k0);
            float4 x0 = xs[0], x1 = xs[1], x2 = xs[2], x3 = xs[3];
            uint4 c0, c1;
            c0.x = pack2_bf16_rne(x0.x, x0.y);
            c0.y = pack2_bf16_rne(x0.z, x0.w);
            c0.z = pack2_bf16_rne(x1.x, x1.y);
            c0.w = pack2_bf16_rne(x1.z, x1.w);
            c1.x = pack2_bf16_rne(x2.x, x2.y);
            c1.y = pack2_bf16_rne(x2.z, x2.w);
            c1.z = pack2_bf16_rne(x3.x, x3.y);
            c1.w = pack2_bf16_rne(x3.z, x3.w);
            ((uint4*)da)[0] = c0;
            ((uint4*)da)[1] = c1;
        }
    };

    stage(0, 0);
    int cur = 0;

    for (int k0 = 0; k0 < IN_F; k0 += BK) {
#if HAVE_ASYNC_LDS
        wait_async0();             // our async writes into buf[cur] done
#endif
        __syncthreads();           // everyone's writes to buf[cur] visible

        if (k0 + BK < IN_F)
            stage(cur ^ 1, k0 + BK);   // overlap next-tile copy with compute

        // ---- load fragments (ISA 16-bit A/B layout: lane=(m|n), halves
        //      pack K = {h*8..h*8+7} then {16+h*8..}), two b128 per frag ----
        const unsigned short* pa = sA[cur];
        const unsigned short* pw = sW[cur];
        FragU a[2], b[4];
        #pragma unroll
        for (int tm = 0; tm < 2; ++tm) {
            const int r = wm * 32 + tm * 16 + lrow;
            const uint4* p = (const uint4*)&pa[r * LDT];
            a[tm].q[0] = p[h];       // halves h*8 .. h*8+7
            a[tm].q[1] = p[2 + h];   // halves 16+h*8 .. 16+h*8+7
        }
        #pragma unroll
        for (int tn = 0; tn < 4; ++tn) {
            const int c = wn * 64 + tn * 16 + lrow;
            const uint4* p = (const uint4*)&pw[c * LDT];
            b[tn].q[0] = p[h];
            b[tn].q[1] = p[2 + h];
        }

        // ---- 8 WMMAs: 32x64 output per wave per K-step ----
        #pragma unroll
        for (int tm = 0; tm < 2; ++tm)
            #pragma unroll
            for (int tn = 0; tn < 4; ++tn)
                acc[tm][tn] = __builtin_amdgcn_wmma_f32_16x16x32_bf16(
                    /*neg_a=*/false, a[tm].v, /*neg_b=*/false, b[tn].v,
                    /*c_mod=*/(short)0, acc[tm][tn],
                    /*reuse_a=*/false, /*reuse_b=*/false);

        cur ^= 1;
    }

    // ---- epilogue: C/D layout -> lane covers col (lrow), rows r + 8*h ----
    #pragma unroll
    for (int tn = 0; tn < 4; ++tn) {
        const int gcol = colBase + wn * 64 + tn * 16 + lrow;
        const float bv = bias[gcol];
        #pragma unroll
        for (int tm = 0; tm < 2; ++tm) {
            const int grow = rowBase + wm * 32 + tm * 16 + h * 8;
            float* o = Out + (size_t)grow * OUT_F + gcol;
            #pragma unroll
            for (int r = 0; r < 8; ++r)
                o[(size_t)r * OUT_F] = acc[tm][tn][r] + bv;
        }
    }
}

// ---------------------------------------------------------------------------
// Launch: X fp32 [8192,4096], weight fp32 [4096,4096], bias fp32 [4096];
// output fp32 [8192,4096].
// d_ws: [0,32MB) = Wb (bf16), [32MB,96MB) = Xb (bf16, if it fits).
// ---------------------------------------------------------------------------
extern "C" void kernel_launch(void* const* d_in, const int* in_sizes, int n_in,
                              void* d_out, int out_size, void* d_ws, size_t ws_size,
                              hipStream_t stream) {
    const float* X    = (const float*)d_in[0];
    const float* W    = (const float*)d_in[1];
    const float* bias = (const float*)d_in[2];
    float* Out        = (float*)d_out;

    const size_t wbBytes = (size_t)OUT_F * IN_F * 2;   // 32 MB
    const size_t xbBytes = (size_t)NTOK  * IN_F * 2;   // 64 MB

    unsigned short* Wb = (unsigned short*)d_ws;
    binarize_weights<<<(OUT_F * IN_F) / (256 * 8), 256, 0, stream>>>(W, (uint4*)Wb);

    dim3 grid(NTOK / BM, OUT_F / BN);
    if (ws_size >= wbBytes + xbBytes) {
        unsigned short* Xb = Wb + (size_t)OUT_F * IN_F;
        convert_x<<<(NTOK * IN_F) / (256 * 8), 256, 0, stream>>>(X, (uint4*)Xb);
        binlinear_gemm<true><<<grid, 256, 0, stream>>>(X, Xb, Wb, bias, Out);
    } else {
        binlinear_gemm<false><<<grid, 256, 0, stream>>>(X, nullptr, Wb, bias, Out);
    }
}
